// MSAAttention_33913061769433
// MI455X (gfx1250) — matrix operand order repaired and checked by reading, not compiled, not run on hardware
//
#include <hip/hip_runtime.h>

// MSA attention with pair bias, MI455X (gfx1250).
// All matmuls on v_wmma_f32_16x16x32_bf16. Every WMMA operand buffer is stored
// in fragment order [tile][lane][16], so fragment loads are 2x b128 per lane.
// Grid dimensions carry the (batch, tile) decomposition so all addressing is
// 32-bit shift/add (no 64-bit div/mod). K4's K-loop is fully unrolled so the
// scheduler overlaps fragment loads with WMMAs without register rotation.

typedef __attribute__((ext_vector_type(16))) __bf16 v16bf;
typedef __attribute__((ext_vector_type(8)))  float  v8f;

#define NSEQ 512
#define NRES 768
#define MSAC 64
#define PAIRC 128
#define NHEAD 8
#define ROWS (NSEQ * NRES)        // 393216 (b,k) rows
#define NCOL (NSEQ * 8)           // 4096 attention-GEMM N dimension
#define KCH  (NRES / 32)          // 24 attention K chunks
#define QT   (NRES / 16)          // 48 q tiles
#define NT   (NCOL / 16)          // 256 n tiles

// ---------- bf16 helpers (native cvt, RNE) ----------
__device__ __forceinline__ __bf16 f2bf(float f) { return (__bf16)f; }
__device__ __forceinline__ float  bf2f(__bf16 b) { return (float)b; }

// ---------- fragment-position maps (contiguous k = 16*half + elem for A and B, 16-bit A 16x32 / B 32x16) ----------
// A: lane = m + 16*(K>>4); elem = K&15
__device__ __forceinline__ int a_idx(int m, int Kl) {
    return (m + ((Kl >> 4) << 4)) * 16 + (Kl & 15);
}
// B: lane = n + 16*(K>>4); elem = K&15
__device__ __forceinline__ int b_idx(int Kl, int n) {
    return (n + ((Kl >> 4) << 4)) * 16 + (Kl & 15);
}
__device__ __forceinline__ v8f bf16_wmma(v16bf a, v16bf b, v8f c) {
    return __builtin_amdgcn_wmma_f32_16x16x32_bf16(false, a, false, b, (short)0, c, false, false);
}
// scatter a layernormed value into A-fragment-ordered LDS (row m, channel c)
__device__ __forceinline__ void stA(__bf16* sh, int m, int c, float v) {
    sh[((c >> 5) << 9) + a_idx(m, c & 31)] = f2bf(v);
}

// ---------- K0a: weights -> bf16 fragment-order B operands ----------
__global__ void k_prep_weights(const float* __restrict__ wpl, const float* __restrict__ vproj,
                               const float* __restrict__ wgate, const float* __restrict__ wout,
                               __bf16* wplF, __bf16* vprojF, __bf16* wgateF, __bf16* woutF) {
    int t = blockIdx.x * blockDim.x + threadIdx.x;
    if (t < 4 * 512) {                 // wplF[kc(4)][lane][16]
        int lane = (t >> 4) & 31, elem = t & 15;
        int n = lane & 15;
        int c = ((t >> 9) << 5) + ((lane >> 4) << 4) + elem;
        wplF[t] = f2bf(n < NHEAD ? wpl[n * PAIRC + c] : 0.0f);
    }
    if (t < 2 * 4 * 512) {             // [kc(2)][nt(4)][lane][16]
        int lane = (t >> 4) & 31, elem = t & 15;
        int nc = (((t >> 9) & 3) << 4) + (lane & 15);
        int Kr = ((t >> 11) << 5) + ((lane >> 4) << 4) + elem;
        vprojF[t] = f2bf(vproj[Kr * MSAC + nc]);        // [i][h*8+c] row-major
        wgateF[t] = f2bf(wgate[nc * MSAC + Kr]);        // transpose for act @ W.T
        woutF[t]  = f2bf(wout[nc * MSAC + Kr]);
    }
}

// ---------- K0b: mask column max -> bf16 ----------
__global__ void k_mask_max(const float* __restrict__ mask, __bf16* mm) {
    int k = blockIdx.x * blockDim.x + threadIdx.x;
    if (k >= NRES) return;
    float m = -3.0e38f;
    for (int s = 0; s < NSEQ; ++s) m = fmaxf(m, mask[s * NRES + k]);
    mm[k] = f2bf(m);
}

// ---------- K1: act LN + v-projection + gate (WMMA), 16 rows / wave ----------
// grid = (ktile 48, batch 512): the 16-row tile never crosses a batch boundary.
__global__ void __launch_bounds__(32)
k_ln_v_gate(const float* __restrict__ act, const float* __restrict__ ascale,
            const float* __restrict__ abias, const __bf16* __restrict__ vprojF,
            const __bf16* __restrict__ wgateF, __bf16* VF, __bf16* Gbuf) {
    __shared__ __align__(32) __bf16 Afrag[2 * 512];
    const int lane = threadIdx.x;
    const int k0 = blockIdx.x * 16;
    const int b_ = blockIdx.y;
    const int rowbase = b_ * NRES + k0;
    if (lane < 16) {
        const float* p = act + (rowbase + lane) * MSAC;
        float s = 0.f, s2 = 0.f;
#pragma unroll
        for (int c = 0; c < MSAC; c += 4) {
            float4 v = *(const float4*)(p + c);
            s  += v.x + v.y + v.z + v.w;
            s2 += v.x * v.x + v.y * v.y + v.z * v.z + v.w * v.w;
        }
        const float mu  = s * (1.f / MSAC);
        const float inv = rsqrtf(s2 * (1.f / MSAC) - mu * mu + 1e-5f);
#pragma unroll
        for (int c = 0; c < MSAC; c += 4) {
            float4 v = *(const float4*)(p + c);
            stA(Afrag, lane, c + 0, (v.x - mu) * inv * ascale[c + 0] + abias[c + 0]);
            stA(Afrag, lane, c + 1, (v.y - mu) * inv * ascale[c + 1] + abias[c + 1]);
            stA(Afrag, lane, c + 2, (v.z - mu) * inv * ascale[c + 2] + abias[c + 2]);
            stA(Afrag, lane, c + 3, (v.w - mu) * inv * ascale[c + 3] + abias[c + 3]);
        }
    }
    __syncthreads();
    const v16bf a0 = *(const v16bf*)&Afrag[0 * 512 + lane * 16];
    const v16bf a1 = *(const v16bf*)&Afrag[1 * 512 + lane * 16];
    const int n = lane & 15, m0 = (lane >> 4) * 8;
    for (int wsel = 0; wsel < 2; ++wsel) {
        const __bf16* WB = wsel ? wgateF : vprojF;
#pragma unroll
        for (int nt = 0; nt < 4; ++nt) {
            v8f acc = {};
            acc = bf16_wmma(a0, *(const v16bf*)&WB[(0 * 4 + nt) * 512 + lane * 16], acc);
            acc = bf16_wmma(a1, *(const v16bf*)&WB[(1 * 4 + nt) * 512 + lane * 16], acc);
            const int col = nt * 16 + n;
            if (!wsel) {                        // v -> VF[h][kc][nt][lane][16] (attn B op)
                const int h = col >> 3, c = col & 7;
                const int ncol = b_ * 8 + c;
                const int tbase = ((h * KCH + (k0 >> 5)) * NT + (ncol >> 4)) << 9;
                const int nl = ncol & 15;
#pragma unroll
                for (int r = 0; r < 8; ++r) {
                    const int k_ = k0 + r + m0;          // k0 is 16-aligned; Kl = k_ & 31
                    VF[(((k_ >> 5) - (k0 >> 5)) * (NT << 9)) + tbase + b_idx(k_ & 31, nl)]
                        = f2bf(acc[r]);
                }
            } else {                            // sigmoid gate -> [row][col] bf16
                const int gbase = rowbase * MSAC + col;
#pragma unroll
                for (int r = 0; r < 8; ++r) {
                    float e = __expf(-acc[r]);
                    Gbuf[gbase + (r + m0) * MSAC] = f2bf(__builtin_amdgcn_rcpf(1.f + e));
                }
            }
        }
    }
}

// ---------- K2: pair LN + logits WMMA (+ mask bias), 16 k-rows / wave ----------
// grid = (ktile 48, q 768)
__global__ void __launch_bounds__(32)
k_pair_logits(const float* __restrict__ pair, const float* __restrict__ pscale,
              const float* __restrict__ pbias, const __bf16* __restrict__ wplF,
              const __bf16* __restrict__ mm, float* Lbuf) {
    __shared__ __align__(32) __bf16 Afrag[4 * 512];
    const int lane = threadIdx.x;
    const int k0 = blockIdx.x * 16;
    const int q  = blockIdx.y;
    if (lane < 16) {
        const float* p = pair + (q * NRES + k0 + lane) * PAIRC;
        float s = 0.f, s2 = 0.f;
#pragma unroll
        for (int c = 0; c < PAIRC; c += 4) {
            float4 v = *(const float4*)(p + c);
            s  += v.x + v.y + v.z + v.w;
            s2 += v.x * v.x + v.y * v.y + v.z * v.z + v.w * v.w;
        }
        const float mu  = s * (1.f / PAIRC);
        const float inv = rsqrtf(s2 * (1.f / PAIRC) - mu * mu + 1e-5f);
#pragma unroll
        for (int c = 0; c < PAIRC; c += 4) {
            float4 v = *(const float4*)(p + c);
            stA(Afrag, lane, c + 0, (v.x - mu) * inv * pscale[c + 0] + pbias[c + 0]);
            stA(Afrag, lane, c + 1, (v.y - mu) * inv * pscale[c + 1] + pbias[c + 1]);
            stA(Afrag, lane, c + 2, (v.z - mu) * inv * pscale[c + 2] + pbias[c + 2]);
            stA(Afrag, lane, c + 3, (v.w - mu) * inv * pscale[c + 3] + pbias[c + 3]);
        }
    }
    __syncthreads();
    v8f acc = {};
#pragma unroll
    for (int kc = 0; kc < 4; ++kc) {
        v16bf a = *(const v16bf*)&Afrag[kc * 512 + lane * 16];
        v16bf b = *(const v16bf*)&wplF[kc * 512 + lane * 16];
        acc = bf16_wmma(a, b, acc);
    }
    const int n = lane & 15, m0 = (lane >> 4) * 8;
    if (n < NHEAD) {
        const int lbase = (n * NRES + q) * NRES + k0 + m0;
#pragma unroll
        for (int r = 0; r < 8; ++r) {
            const float bias = 1e9f * (bf2f(mm[k0 + m0 + r]) - 1.0f);  // bf16 round-trip like ref
            Lbuf[lbase + r] = acc[r] + bias;
        }
    }
}

// ---------- K3: softmax over k, write weights in A-fragment order ----------
// grid = (q 768, h 8)
__global__ void __launch_bounds__(256)
k_softmax(const float* __restrict__ Lbuf, __bf16* WF) {
    __shared__ float red[256];
    const int q = blockIdx.x, h = blockIdx.y;
    const int qt = q >> 4, m = q & 15;
    const float* row = Lbuf + (h * NRES + q) * NRES;
    __bf16* wtile = WF + (((h * QT + qt) * KCH) << 9);
    const int t = threadIdx.x;
    float mx = -3.0e38f;
    for (int k = t; k < NRES; k += 256) mx = fmaxf(mx, row[k]);
    red[t] = mx; __syncthreads();
    for (int s = 128; s > 0; s >>= 1) { if (t < s) red[t] = fmaxf(red[t], red[t + s]); __syncthreads(); }
    mx = red[0]; __syncthreads();
    float sum = 0.f;
    for (int k = t; k < NRES; k += 256) sum += __expf(row[k] - mx);
    red[t] = sum; __syncthreads();
    for (int s = 128; s > 0; s >>= 1) { if (t < s) red[t] += red[t + s]; __syncthreads(); }
    const float inv = __builtin_amdgcn_rcpf(red[0]);
    for (int k = t; k < NRES; k += 256)
        wtile[((k >> 5) << 9) + a_idx(m, k & 31)] = f2bf(__expf(row[k] - mx) * inv);
}

// ---------- K4: attention GEMM per (h, q-tile, 4 n-tiles), fused gate ----------
// grid = (ngroup 64, qtile 48, h 8); fully unrolled K-loop for load/WMMA overlap.
__global__ void __launch_bounds__(32)
k_attn(const __bf16* __restrict__ WF, const __bf16* __restrict__ VF,
       const __bf16* __restrict__ Gbuf, __bf16* VGF) {
    const int lane = threadIdx.x;
    const int ng = blockIdx.x;
    const int qt = blockIdx.y;
    const int h  = blockIdx.z;
    const __bf16* Ap = WF + (((h * QT + qt) * KCH) << 9) + lane * 16;
    const __bf16* Bp = VF + ((h * KCH * NT + ng * 4) << 9) + lane * 16;

    v8f acc0 = {}, acc1 = {}, acc2 = {}, acc3 = {};
#pragma unroll
    for (int kc = 0; kc < KCH; ++kc) {
        const v16bf a  = *(const v16bf*)(Ap + (kc << 9));
        const v16bf b0 = *(const v16bf*)(Bp + ((kc * NT + 0) << 9));
        const v16bf b1 = *(const v16bf*)(Bp + ((kc * NT + 1) << 9));
        const v16bf b2 = *(const v16bf*)(Bp + ((kc * NT + 2) << 9));
        const v16bf b3 = *(const v16bf*)(Bp + ((kc * NT + 3) << 9));
        acc0 = bf16_wmma(a, b0, acc0);
        acc1 = bf16_wmma(a, b1, acc1);
        acc2 = bf16_wmma(a, b2, acc2);
        acc3 = bf16_wmma(a, b3, acc3);
    }
    const int n = lane & 15, m0 = (lane >> 4) * 8;
    v8f* accs[4] = { &acc0, &acc1, &acc2, &acc3 };
#pragma unroll
    for (int s = 0; s < 4; ++s) {
        const int nf = (ng * 4 + s) * 16 + n;
        const int b_ = nf >> 3;
        const int c  = nf & 7;
        const int col = h * 8 + c, kc2 = col >> 5, Kl2 = col & 31;
        const int rt0 = b_ * QT + qt;                 // (row >> 4), row = b_*768 + q
        const int gbase = (b_ * NRES + qt * 16) * MSAC + col;
        const int vbase = ((rt0 * 2 + kc2) << 9);
#pragma unroll
        for (int r = 0; r < 8; ++r) {
            const int m2 = r + m0;                    // row & 15
            const float g = bf2f(Gbuf[gbase + m2 * MSAC]);
            VGF[vbase + a_idx(m2, Kl2)] = f2bf((*accs[s])[r] * g);
        }
    }
}

// ---------- K5: output projection -> f32 out ----------
__global__ void __launch_bounds__(32)
k_outproj(const __bf16* __restrict__ VGF, const __bf16* __restrict__ woutF,
          float* __restrict__ out) {
    const int lane = threadIdx.x;
    const int rt = blockIdx.x;
    const v16bf a0 = *(const v16bf*)&VGF[((rt * 2 + 0) << 9) + lane * 16];
    const v16bf a1 = *(const v16bf*)&VGF[((rt * 2 + 1) << 9) + lane * 16];
    const int n = lane & 15, m0 = (lane >> 4) * 8;
#pragma unroll
    for (int nt = 0; nt < 4; ++nt) {
        v8f acc = {};
        acc = bf16_wmma(a0, *(const v16bf*)&woutF[(0 * 4 + nt) * 512 + lane * 16], acc);
        acc = bf16_wmma(a1, *(const v16bf*)&woutF[(1 * 4 + nt) * 512 + lane * 16], acc);
        const int obase = (rt * 16 + m0) * MSAC + nt * 16 + n;
#pragma unroll
        for (int r = 0; r < 8; ++r) out[obase + r * MSAC] = acc[r];
    }
}

// ---------- host launch ----------
static inline size_t align256(size_t x) { return (x + 255) & ~(size_t)255; }

extern "C" void kernel_launch(void* const* d_in, const int* in_sizes, int n_in,
                              void* d_out, int out_size, void* d_ws, size_t ws_size,
                              hipStream_t stream) {
    const float* act    = (const float*)d_in[0];
    const float* mask   = (const float*)d_in[1];
    const float* pair   = (const float*)d_in[2];
    const float* ascale = (const float*)d_in[3];
    const float* abias  = (const float*)d_in[4];
    const float* pscale = (const float*)d_in[5];
    const float* pbias  = (const float*)d_in[6];
    const float* wpl    = (const float*)d_in[7];
    const float* vproj  = (const float*)d_in[8];
    const float* wgate  = (const float*)d_in[9];
    const float* wout   = (const float*)d_in[10];
    float* out = (float*)d_out;
    char* ws = (char*)d_ws;

    size_t o = 0;
    __bf16* wplF   = (__bf16*)(ws + o); o = align256(o + (size_t)4 * 512 * 2);
    __bf16* vprojF = (__bf16*)(ws + o); o = align256(o + (size_t)8 * 512 * 2);
    __bf16* wgateF = (__bf16*)(ws + o); o = align256(o + (size_t)8 * 512 * 2);
    __bf16* woutF  = (__bf16*)(ws + o); o = align256(o + (size_t)8 * 512 * 2);
    __bf16* mm     = (__bf16*)(ws + o); o = align256(o + (size_t)NRES * 2);
    __bf16* VF     = (__bf16*)(ws + o); o = align256(o + (size_t)NHEAD * KCH * NT * 512 * 2);
    __bf16* Gbuf   = (__bf16*)(ws + o); o = align256(o + (size_t)ROWS * MSAC * 2);
    float*  Lbuf   = (float*)(ws + o);  o = align256(o + (size_t)NHEAD * NRES * NRES * 4);
    __bf16* WF     = (__bf16*)(ws + o); o = align256(o + (size_t)NHEAD * QT * KCH * 512 * 2);
    __bf16* VGF    = (__bf16*)(ws + o); o = align256(o + (size_t)(ROWS / 16) * 2 * 512 * 2);
    (void)ws_size; (void)in_sizes; (void)n_in; (void)out_size;

    k_prep_weights<<<16, 256, 0, stream>>>(wpl, vproj, wgate, wout,
                                           wplF, vprojF, wgateF, woutF);
    k_mask_max<<<3, 256, 0, stream>>>(mask, mm);
    k_ln_v_gate<<<dim3(QT, NSEQ), 32, 0, stream>>>(act, ascale, abias, vprojF, wgateF, VF, Gbuf);
    k_pair_logits<<<dim3(QT, NRES), 32, 0, stream>>>(pair, pscale, pbias, wplF, mm, Lbuf);
    k_softmax<<<dim3(NRES, NHEAD), 256, 0, stream>>>(Lbuf, WF);
    k_attn<<<dim3(64, QT, NHEAD), 32, 0, stream>>>(WF, VF, Gbuf, VGF);
    k_outproj<<<ROWS / 16, 32, 0, stream>>>(VGF, woutF, out);
}
